// GroupedQueryAttention_65687229825440
// MI455X (gfx1250) — compile-verified, matrix operand order checked by reading, not run on hardware
//
#include <hip/hip_runtime.h>
#include <hip/hip_bf16.h>

#define TT 4096
#define DM 1024
#define NQ 16
#define NKV 4
#define HD 64

typedef __attribute__((ext_vector_type(16))) __bf16 v16bf;
typedef __attribute__((ext_vector_type(8)))  float  v8f;
typedef __attribute__((ext_vector_type(4)))  unsigned int v4u;
typedef __attribute__((ext_vector_type(8)))  unsigned int v8u;

typedef unsigned short u16t;
typedef unsigned long long u64t;

__device__ inline u16t f2bf(float f) {
  unsigned int u = __float_as_uint(f);
  unsigned int r = (u + 0x7FFFu + ((u >> 16) & 1u)) >> 16;
  return (u16t)r;
}

union FragB {
  v16bf v;
  unsigned int w[8];
};

// Load a 16x32 bf16 fragment in the CDNA5 WMMA A/B register layout from
// row-major LDS (row stride `stride` in elements). Per ISA 7.12.2:
//   lanes 0-15 : rows 0-15, VGPR i holds K = {2i, 2i+1} (i<4) / {16+2(i-4), ...}
//   lanes 16-31: same rows, K offset +8
__device__ inline v16bf ld_frag(const u16t* lds, int stride, int rbase, int kbase) {
  const int lane = threadIdx.x & 31;
  const int half = lane >> 4;
  const int r    = lane & 15;
  FragB f;
#pragma unroll
  for (int i = 0; i < 8; ++i) {
    int k = kbase + ((i & 3) * 2) + ((i >> 2) * 16) + half * 8;
    f.w[i] = *(const unsigned int*)&lds[(rbase + r) * stride + k];
  }
  return f.v;
}

// ---- Tensor Data Mover: 2D bf16 tile Global -> LDS ----------------------
// D# per CDNA5 ISA. Group0: count=1, lds_addr,
// global_addr (tile start), type=2. Group1: data_size=1 (2 bytes),
// tensor_dim0/1, tile_dim0/1 (tile rows packed contiguously in LDS,
// so LDS layout is row-major with stride == tile_dim0), dim0 stride.
// Groups 2/3 zero (<=2D tensor). Issued by one wave; EXEC is ignored.
__device__ inline void tdm_load_2d(unsigned lds_addr, const void* gptr,
                                   unsigned td0, unsigned td1,
                                   unsigned tl0, unsigned tl1,
                                   unsigned stride0) {
  u64t ga = (u64t)gptr;
  v4u g0;
  g0[0] = 1u;                                                   // count=1
  g0[1] = lds_addr;                                             // lds_addr[63:32]
  g0[2] = (unsigned)ga;                                         // global_addr[31:0]
  g0[3] = (unsigned)((ga >> 32) & 0x01FFFFFFu) | (2u << 30);    // [56:32] | type=2
  v8u g1;
  g1[0] = (1u << 16);                                           // data_size=2B
  g1[1] = (td0 & 0xFFFFu) << 16;                                // tensor_dim0[15:0]
  g1[2] = ((td0 >> 16) & 0xFFFFu) | ((td1 & 0xFFFFu) << 16);    // d0 hi | d1 lo
  g1[3] = ((td1 >> 16) & 0xFFFFu) | ((tl0 & 0xFFFFu) << 16);    // d1 hi | tile_dim0
  g1[4] = (tl1 & 0xFFFFu);                                      // tile_dim1 (dim2=0)
  g1[5] = stride0;                                              // dim0_stride[31:0]
  g1[6] = 0u;
  g1[7] = 0u;
  v4u gz = {0u, 0u, 0u, 0u};
  asm volatile("tensor_load_to_lds %0, %1, %2, %3"
               :: "s"(g0), "s"(g1), "s"(gz), "s"(gz)
               : "memory");
}

__device__ inline void tdm_wait() {
#if __has_builtin(__builtin_amdgcn_s_wait_tensorcnt)
  __builtin_amdgcn_s_wait_tensorcnt(0);
#else
  asm volatile("s_wait_tensorcnt 0x0" ::: "memory");
#endif
}

__global__ void cvt_f32_bf16(const float* __restrict__ in, u16t* __restrict__ out, int n) {
  int i = blockIdx.x * blockDim.x + threadIdx.x;
  if (i < n) out[i] = f2bf(in[i]);
}

// C = A(MxK) @ B(KxN), bf16 inputs, fp32 accumulate via v_wmma_f32_16x16x32_bf16.
// 64x64 tile per 128-thread block (4 waves, each owning a 16-row stripe).
// A tile is DMA'd by the TDM; B tile is transposed manually into (n, k).
__global__ __launch_bounds__(128) void gemm_bf16(
    const u16t* __restrict__ A, const u16t* __restrict__ B,
    float* __restrict__ outF, u16t* __restrict__ outB,
    int M, int N, int K, float scale)
{
  __shared__ u16t a_lds[64 * 32];  // row-major (m, k) — TDM destination
  __shared__ u16t b_lds[64 * 32];  // col-major (n, k)

  const int tid  = threadIdx.x;
  const int wave = tid >> 5;
  const int lane = tid & 31;
  const int half = lane >> 4;
  const int nc   = lane & 15;
  const int mBase = blockIdx.y * 64;
  const int nBase = blockIdx.x * 64;
  const unsigned a_lds_addr = (unsigned)(u64t)a_lds;

  v8f vzero = {0.f, 0.f, 0.f, 0.f, 0.f, 0.f, 0.f, 0.f};
  v8f acc[4];
#pragma unroll
  for (int t = 0; t < 4; ++t) acc[t] = vzero;

  for (int k0 = 0; k0 < K; k0 += 32) {
    __syncthreads();
    // A tile 64x32 via Tensor Data Mover (rows packed -> stride 32 in LDS)
    if (wave == 0)
      tdm_load_2d(a_lds_addr, &A[(u64t)mBase * K + k0],
                  (unsigned)K, (unsigned)M, 32u, 64u, (unsigned)K);
    // B tile 32x64, transposed into (n, k) so K-pairs are contiguous
    for (int i = tid; i < 32 * 64; i += 128) {
      int k = i >> 6, n = i & 63;
      b_lds[n * 32 + k] = B[(u64t)(k0 + k) * N + nBase + n];
    }
    if (k0 + 32 < K)
      __builtin_prefetch(&A[(u64t)(mBase + (tid >> 3)) * K + k0 + 32], 0, 1);
    if (wave == 0) tdm_wait();
    __syncthreads();

    v16bf aF = ld_frag(a_lds, 32, wave * 16, 0);
#pragma unroll
    for (int t = 0; t < 4; ++t) {
      v16bf bF = ld_frag(b_lds, 32, t * 16, 0);
      acc[t] = __builtin_amdgcn_wmma_f32_16x16x32_bf16(
          false, aF, false, bF, (short)0, acc[t], false, false);
    }
  }

  // Epilogue: C/D layout -> VGPR r holds M=r (lanes 0-15) / M=r+8 (lanes 16-31)
#pragma unroll
  for (int t = 0; t < 4; ++t) {
#pragma unroll
    for (int r = 0; r < 8; ++r) {
      int Mg = mBase + wave * 16 + r + half * 8;
      int Ng = nBase + t * 16 + nc;
      float val = acc[t][r] * scale;
      if (outF) outF[(u64t)Mg * N + Ng] = val;
      else      outB[(u64t)Mg * N + Ng] = f2bf(val);
    }
  }
}

// Flash attention, causal, GQA (q head h -> kv head h/4).
// Grid: (T/64 q-blocks, NQ heads). Block: 128 threads = 4 waves; wave w owns
// query rows [qb*64 + 16w, +16). Q pre-scaled by 1/sqrt(HD) in projection.
// Q block and K tiles arrive via the Tensor Data Mover; V is transposed manually.
__global__ __launch_bounds__(128) void gqa_attn(
    const u16t* __restrict__ Qg, const u16t* __restrict__ Kg,
    const u16t* __restrict__ Vg, u16t* __restrict__ Og)
{
  __shared__ u16t k_lds[64 * 64];      // (key, d) row-major; also Q staging; TDM dest
  __shared__ u16t v_lds[64 * 64];      // (d, key) transposed
  __shared__ u16t p_lds[4][16 * 64];   // per-wave P tile (row, key)

  const int tid  = threadIdx.x;
  const int wave = tid >> 5;
  const int lane = tid & 31;
  const int half = lane >> 4;
  const int nc   = lane & 15;
  const int qb   = blockIdx.x;
  const int h    = blockIdx.y;
  const int kvh  = h >> 2;
  const int qRowBase = qb * 64 + wave * 16;
  const unsigned k_lds_addr = (unsigned)(u64t)k_lds;

  // Stage Q block (64 x HD) through LDS via TDM, pull per-wave A fragments
  if (wave == 0) {
    tdm_load_2d(k_lds_addr, &Qg[(u64t)(qb * 64) * (NQ * HD) + h * HD],
                NQ * HD, TT, 64u, 64u, NQ * HD);
    tdm_wait();
  }
  __syncthreads();
  v16bf qA0 = ld_frag(k_lds, 64, wave * 16, 0);
  v16bf qA1 = ld_frag(k_lds, 64, wave * 16, 32);
  __syncthreads();

  v8f vzero = {0.f, 0.f, 0.f, 0.f, 0.f, 0.f, 0.f, 0.f};
  v8f oacc[4];
#pragma unroll
  for (int t = 0; t < 4; ++t) oacc[t] = vzero;
  float mrow[8], lrow[8];
#pragma unroll
  for (int r = 0; r < 8; ++r) { mrow[r] = -1e30f; lrow[r] = 0.f; }

  for (int j = 0; j <= qb; ++j) {
    const int t0 = j * 64;
    // K tile (key, d) via TDM; V tile transposed to (d, key) manually so the
    // PV B-fragments read contiguous key pairs.
    if (wave == 0)
      tdm_load_2d(k_lds_addr, &Kg[(u64t)t0 * (NKV * HD) + kvh * HD],
                  NKV * HD, TT, 64u, 64u, NKV * HD);
    for (int i = tid; i < 64 * 64; i += 128) {
      int key = i >> 6, d = i & 63;
      v_lds[d * 64 + key] = Vg[(u64t)(t0 + key) * (NKV * HD) + kvh * HD + d];
    }
    if (j < qb)
      __builtin_prefetch(&Kg[(u64t)(t0 + 64 + (tid >> 3)) * (NKV * HD) + kvh * HD], 0, 1);
    if (wave == 0) tdm_wait();
    __syncthreads();

    // S = Q @ K^T : B fragment n-index = key, k-index = d
    v8f s[4];
#pragma unroll
    for (int t = 0; t < 4; ++t) {
      v8f z = vzero;
      v16bf b0 = ld_frag(k_lds, 64, t * 16, 0);
      z = __builtin_amdgcn_wmma_f32_16x16x32_bf16(false, qA0, false, b0, (short)0, z, false, false);
      v16bf b1 = ld_frag(k_lds, 64, t * 16, 32);
      z = __builtin_amdgcn_wmma_f32_16x16x32_bf16(false, qA1, false, b1, (short)0, z, false, false);
      s[t] = z;
    }

    // Causal mask on the diagonal block
    if (j == qb) {
#pragma unroll
      for (int t = 0; t < 4; ++t) {
        int colg = t0 + t * 16 + nc;
#pragma unroll
        for (int r = 0; r < 8; ++r) {
          int rowg = qRowBase + r + half * 8;
          if (colg > rowg) s[t][r] = -1e30f;
        }
      }
    }

    // Online softmax: rows live across 16-lane groups (shfl_xor 1/2/4/8)
#pragma unroll
    for (int r = 0; r < 8; ++r) {
      float v = fmaxf(fmaxf(s[0][r], s[1][r]), fmaxf(s[2][r], s[3][r]));
      v = fmaxf(v, __shfl_xor(v, 1));
      v = fmaxf(v, __shfl_xor(v, 2));
      v = fmaxf(v, __shfl_xor(v, 4));
      v = fmaxf(v, __shfl_xor(v, 8));
      float mnew = fmaxf(mrow[r], v);
      float corr = __expf(mrow[r] - mnew);
      float p0 = __expf(s[0][r] - mnew);
      float p1 = __expf(s[1][r] - mnew);
      float p2 = __expf(s[2][r] - mnew);
      float p3 = __expf(s[3][r] - mnew);
      s[0][r] = p0; s[1][r] = p1; s[2][r] = p2; s[3][r] = p3;
      float rs = p0 + p1 + p2 + p3;
      rs += __shfl_xor(rs, 1);
      rs += __shfl_xor(rs, 2);
      rs += __shfl_xor(rs, 4);
      rs += __shfl_xor(rs, 8);
      lrow[r] = lrow[r] * corr + rs;
      mrow[r] = mnew;
#pragma unroll
      for (int t = 0; t < 4; ++t) oacc[t][r] *= corr;
    }

    // P (C/D layout) -> LDS (row, key) so it can be reloaded as an A fragment
#pragma unroll
    for (int t = 0; t < 4; ++t) {
#pragma unroll
      for (int r = 0; r < 8; ++r) {
        p_lds[wave][(r + half * 8) * 64 + t * 16 + nc] = f2bf(s[t][r]);
      }
    }
    __syncthreads();

    // O += P @ V : A = P rows, B fragment n-index = d, k-index = key
    v16bf pA0 = ld_frag(p_lds[wave], 64, 0, 0);
    v16bf pA1 = ld_frag(p_lds[wave], 64, 0, 32);
#pragma unroll
    for (int t = 0; t < 4; ++t) {
      v16bf b0 = ld_frag(v_lds, 64, t * 16, 0);
      oacc[t] = __builtin_amdgcn_wmma_f32_16x16x32_bf16(false, pA0, false, b0, (short)0, oacc[t], false, false);
      v16bf b1 = ld_frag(v_lds, 64, t * 16, 32);
      oacc[t] = __builtin_amdgcn_wmma_f32_16x16x32_bf16(false, pA1, false, b1, (short)0, oacc[t], false, false);
    }
    __syncthreads();
  }

  // Normalize and emit bf16 attention output (T x NQ*HD)
#pragma unroll
  for (int t = 0; t < 4; ++t) {
#pragma unroll
    for (int r = 0; r < 8; ++r) {
      int rowg = qRowBase + r + half * 8;
      int colg = h * HD + t * 16 + nc;
      Og[(u64t)rowg * (NQ * HD) + colg] = f2bf(oacc[t][r] / lrow[r]);
    }
  }
}

extern "C" void kernel_launch(void* const* d_in, const int* in_sizes, int n_in,
                              void* d_out, int out_size, void* d_ws, size_t ws_size,
                              hipStream_t stream) {
  const float* x  = (const float*)d_in[0];
  const float* Wq = (const float*)d_in[1];
  const float* Wk = (const float*)d_in[2];
  const float* Wv = (const float*)d_in[3];
  const float* Wo = (const float*)d_in[4];
  float* out = (float*)d_out;

  // Workspace layout (bf16 tensors), ~34.6 MB total
  u16t* xb  = (u16t*)d_ws;                        // TT x DM
  u16t* wqb = xb  + (size_t)TT * DM;              // DM x DM
  u16t* wkb = wqb + (size_t)DM * DM;              // DM x (NKV*HD)
  u16t* wvb = wkb + (size_t)DM * (NKV * HD);      // DM x (NKV*HD)
  u16t* wob = wvb + (size_t)DM * (NKV * HD);      // (NQ*HD) x DM
  u16t* qws = wob + (size_t)(NQ * HD) * DM;       // TT x (NQ*HD)
  u16t* kws = qws + (size_t)TT * (NQ * HD);       // TT x (NKV*HD)
  u16t* vws = kws + (size_t)TT * (NKV * HD);      // TT x (NKV*HD)
  u16t* aws = vws + (size_t)TT * (NKV * HD);      // TT x (NQ*HD)

  cvt_f32_bf16<<<(TT * DM + 255) / 256, 256, 0, stream>>>(x, xb, TT * DM);
  cvt_f32_bf16<<<(DM * DM + 255) / 256, 256, 0, stream>>>(Wq, wqb, DM * DM);
  cvt_f32_bf16<<<(DM * NKV * HD + 255) / 256, 256, 0, stream>>>(Wk, wkb, DM * NKV * HD);
  cvt_f32_bf16<<<(DM * NKV * HD + 255) / 256, 256, 0, stream>>>(Wv, wvb, DM * NKV * HD);
  cvt_f32_bf16<<<(NQ * HD * DM + 255) / 256, 256, 0, stream>>>(Wo, wob, NQ * HD * DM);

  dim3 blk(128, 1, 1);
  // Q = (x @ Wq) * 1/sqrt(HD); K = x @ Wk; V = x @ Wv   (bf16 outputs)
  gemm_bf16<<<dim3(DM / 64, TT / 64), blk, 0, stream>>>(xb, wqb, nullptr, qws, TT, DM, DM, 0.125f);
  gemm_bf16<<<dim3((NKV * HD) / 64, TT / 64), blk, 0, stream>>>(xb, wkb, nullptr, kws, TT, NKV * HD, DM, 1.0f);
  gemm_bf16<<<dim3((NKV * HD) / 64, TT / 64), blk, 0, stream>>>(xb, wvb, nullptr, vws, TT, NKV * HD, DM, 1.0f);

  // Causal GQA flash attention
  gqa_attn<<<dim3(TT / 64, NQ), blk, 0, stream>>>(qws, kws, vws, aws);

  // out = attn @ Wo  (fp32 output)
  gemm_bf16<<<dim3(DM / 64, TT / 64), blk, 0, stream>>>(aws, wob, out, nullptr, TT, DM, DM, 1.0f);
}
